// GATConvLayer_72404558676395
// MI455X (gfx1250) — matrix operand order repaired and checked by reading, not compiled, not run on hardware
//
#include <hip/hip_runtime.h>
#include <hip/hip_bf16.h>

// ---------------------------------------------------------------------------
// GAT layer (dense attention), MI455X / gfx1250, wave32 + WMMA bf16.
//   B=2, N=4096, IN=128, H=4, D=64, HD=256
// ---------------------------------------------------------------------------

#define GAT_B   2
#define GAT_N   4096
#define GAT_IN  128
#define GAT_H   4
#define GAT_D   64
#define GAT_HD  256            // H*D
#define GAT_M   (GAT_B * GAT_N) // 8192 rows for projection
#define SLOPE   0.2f

typedef __attribute__((ext_vector_type(16))) __bf16 v16bf;
typedef __attribute__((ext_vector_type(8)))  float  v8f;

union AFrag {                   // 16-bit A/B matrix fragment: 8 VGPRs / lane
    v16bf    v;
    __bf16   e[16];
    uint32_t u[8];
    uint4    q[2];
};

// ---------------------------------------------------------------------------
// Pass 0a: X (f32) -> Xbf (bf16), row-major [M][128]
__global__ __launch_bounds__(256) void gat_cvt_x(const float* __restrict__ X,
                                                 __bf16* __restrict__ Xbf) {
    int i = blockIdx.x * 256 + threadIdx.x;      // 0 .. M*128-1
    Xbf[i] = (__bf16)X[i];
}

// Pass 0b: W (f32 [128][256]) -> WT (bf16 [256][128])  (K contiguous)
__global__ __launch_bounds__(256) void gat_cvt_w(const float* __restrict__ W,
                                                 __bf16* __restrict__ WT) {
    int i = blockIdx.x * 256 + threadIdx.x;      // 0 .. 256*128-1
    int n = i >> 7, k = i & 127;
    WT[i] = (__bf16)W[k * GAT_HD + n];
}

// ---------------------------------------------------------------------------
// Pass 1: projection h = X*W via v_wmma_f32_16x16x32_bf16.
// One wave -> 16 rows x 64 cols (one head). 2048 waves total.
// Writes hF  (f32, [M][256])  and hT (bf16, [B*H][64][N]  d-major).
__global__ __launch_bounds__(256) void gat_proj(const __bf16* __restrict__ Xbf,
                                                const __bf16* __restrict__ WT,
                                                float* __restrict__ hF,
                                                __bf16* __restrict__ hT) {
    const int tid  = threadIdx.x;
    const int wave = tid >> 5;
    const int lane = tid & 31;
    const int half = lane >> 4;
    const int l16  = lane & 15;

    const int wid   = blockIdx.x * 8 + wave;   // 0..2047
    const int head  = wid & 3;
    const int mtile = wid >> 2;                // 0..511
    const int m0    = mtile * 16;              // row base (global, 0..8191)

    v8f acc[4] = {v8f{}, v8f{}, v8f{}, v8f{}};

    #pragma unroll
    for (int k0 = 0; k0 < GAT_IN; k0 += 32) {
        // A fragment: row = m0+l16 ; K = k0 + 8*half + {0..7}, k0+16+8*half + {0..7}
        AFrag a;
        const __bf16* arow = Xbf + (size_t)(m0 + l16) * GAT_IN + k0 + 8 * half;
        a.q[0] = *(const uint4*)(arow);
        a.q[1] = *(const uint4*)(arow + 16);

        #pragma unroll
        for (int t = 0; t < 4; ++t) {
            // B fragment: col n = head*64 + t*16 + l16 ; K = k0 + 8*half + {0..7}, k0+16+8*half + {0..7}
            AFrag bf;
            const __bf16* brow = WT + (size_t)(head * 64 + t * 16 + l16) * GAT_IN
                                    + k0 + 8 * half;
            bf.q[0] = *(const uint4*)(brow);
            bf.q[1] = *(const uint4*)(brow + 16);
            acc[t] = __builtin_amdgcn_wmma_f32_16x16x32_bf16(
                         false, a.v, false, bf.v, (short)0, acc[t], false, false);
        }
    }

    // C layout: lane l: N = l16, M = r + 8*half
    const int b = m0 >> 12;                 // 4096 rows per batch
    const int nbase = (m0 & 4095);
    #pragma unroll
    for (int t = 0; t < 4; ++t) {
        const int d  = t * 16 + l16;
        const int hd = head * 64 + d;
        #pragma unroll
        for (int r = 0; r < 8; ++r) {
            const int   row = m0 + r + 8 * half;
            const float val = acc[t][r];
            hF[(size_t)row * GAT_HD + hd] = val;
            hT[((size_t)(b * GAT_H + head) * GAT_D + d) * GAT_N
               + nbase + r + 8 * half] = (__bf16)val;
        }
    }
}

// ---------------------------------------------------------------------------
// Pass 2: per-node attention scalars e_src/e_dst (exact, from f32 h)
__global__ __launch_bounds__(256) void gat_escore(const float* __restrict__ hF,
                                                  const float* __restrict__ a_src,
                                                  const float* __restrict__ a_dst,
                                                  float* __restrict__ e_src,
                                                  float* __restrict__ e_dst) {
    int idx = blockIdx.x * 256 + threadIdx.x;  // 0 .. B*H*N-1
    int bh = idx >> 12, n = idx & 4095;
    int b = bh >> 2, head = bh & 3;
    const float* hp = hF + ((size_t)(b * GAT_N + n)) * GAT_HD + head * GAT_D;
    const float* as = a_src + head * GAT_D;
    const float* ad = a_dst + head * GAT_D;
    float ss = 0.f, dd = 0.f;
    #pragma unroll 8
    for (int d = 0; d < GAT_D; ++d) {
        float v = hp[d];
        ss = fmaf(v, as[d], ss);
        dd = fmaf(v, ad[d], dd);
    }
    e_src[idx] = ss;
    e_dst[idx] = dd;
}

// Pass 2b: dmax[bh] = max_j e_dst[bh][j]   (LDS tree reduce)
__global__ __launch_bounds__(256) void gat_dmax(const float* __restrict__ e_dst,
                                                float* __restrict__ dmax) {
    __shared__ float sm[256];
    const int bh = blockIdx.x, t = threadIdx.x;
    float m = -3.0e38f;
    for (int i = t; i < GAT_N; i += 256) m = fmaxf(m, e_dst[bh * GAT_N + i]);
    sm[t] = m;
    __syncthreads();
    for (int s = 128; s > 0; s >>= 1) {
        if (t < s) sm[t] = fmaxf(sm[t], sm[t + s]);
        __syncthreads();
    }
    if (t == 0) dmax[bh] = sm[0];
}

// ---------------------------------------------------------------------------
// Pass 3: flash-style attention: out = softmax(lr(s_i + d_j)) @ h, + bias, ELU.
// Row max is closed-form: m_i = lr(s_i + dmax)  (monotonicity of leakyrelu).
// One wave -> 16 rows x full D=64 for one (b,head). 2048 waves.
__global__ __launch_bounds__(256) void gat_flash(const float* __restrict__ e_src,
                                                 const float* __restrict__ e_dst,
                                                 const float* __restrict__ dmax,
                                                 const __bf16* __restrict__ hT,
                                                 const float* __restrict__ bias,
                                                 float* __restrict__ out) {
    const int tid  = threadIdx.x;
    const int wave = tid >> 5;
    const int lane = tid & 31;
    const int half = lane >> 4;
    const int l16  = lane & 15;

    const int bh     = blockIdx.x >> 5;      // 0..7
    const int rowblk = blockIdx.x & 31;      // 0..31  (128 rows each)
    const int m0     = rowblk * 128 + wave * 16;
    const int b = bh >> 2, head = bh & 3;

    const float*  esrc = e_src + (size_t)bh * GAT_N;
    const float*  edst = e_dst + (size_t)bh * GAT_N;
    const __bf16* hB   = hT + (size_t)bh * GAT_D * GAT_N;

    const float dm = dmax[bh];
    const float s  = esrc[m0 + l16];                 // A layout: lane row = l16
    float xm = s + dm;
    xm = (xm > 0.f) ? xm : SLOPE * xm;               // row max of lr scores

    v8f   acc[4] = {v8f{}, v8f{}, v8f{}, v8f{}};
    float psum = 0.f;

    for (int j0 = 0; j0 < GAT_N; j0 += 32) {
        __builtin_prefetch(edst + j0 + 256, 0, 1);   // global_prefetch_b8

        // 16 e_dst values owned by this lane (A layout K split by half-wave)
        const float4 dA = *(const float4*)(edst + j0 + 8 * half);
        const float4 dB = *(const float4*)(edst + j0 + 8 * half + 4);
        const float4 dC = *(const float4*)(edst + j0 + 16 + 8 * half);
        const float4 dD = *(const float4*)(edst + j0 + 16 + 8 * half + 4);
        const float dv[16] = {dA.x, dA.y, dA.z, dA.w, dB.x, dB.y, dB.z, dB.w,
                              dC.x, dC.y, dC.z, dC.w, dD.x, dD.y, dD.z, dD.w};

        AFrag a;
        #pragma unroll
        for (int q = 0; q < 16; ++q) {
            float x = s + dv[q];
            x = (x > 0.f) ? x : SLOPE * x;           // leaky_relu
            float p = __expf(x - xm);                // unnormalized softmax
            psum += p;
            a.e[q] = (__bf16)p;
        }

        // B fragments from hT (d-major): lane col d = t*16+l16, K = j0+8*half+{0..7}, j0+16+8*half+{0..7}
        const int jb = j0 + 8 * half;
        #pragma unroll
        for (int t = 0; t < 4; ++t) {
            AFrag bf;
            const __bf16* brow = hB + (size_t)(t * 16 + l16) * GAT_N + jb;
            bf.q[0] = *(const uint4*)(brow);
            bf.q[1] = *(const uint4*)(brow + 16);
            acc[t] = __builtin_amdgcn_wmma_f32_16x16x32_bf16(
                         false, a.v, false, bf.v, (short)0, acc[t], false, false);
        }
    }

    // full row sums: lanes l and l+16 hold complementary K halves of row l16
    psum += __shfl_xor(psum, 16, 32);
    float inv[8];
    #pragma unroll
    for (int r = 0; r < 8; ++r)
        inv[r] = 1.0f / __shfl(psum, r + 8 * half, 32);  // row = r + 8*half

    const float* bi = bias + head * GAT_D;
    float* ob = out + ((size_t)b * GAT_N + m0) * GAT_HD + head * GAT_D;
    #pragma unroll
    for (int t = 0; t < 4; ++t) {
        const int d  = t * 16 + l16;
        const float bv = bi[d];
        #pragma unroll
        for (int r = 0; r < 8; ++r) {
            float val = acc[t][r] * inv[r] + bv;
            val = (val > 0.f) ? val : (__expf(val) - 1.0f);  // ELU
            ob[(size_t)(r + 8 * half) * GAT_HD + d] = val;
        }
    }
}

// ---------------------------------------------------------------------------
extern "C" void kernel_launch(void* const* d_in, const int* in_sizes, int n_in,
                              void* d_out, int out_size, void* d_ws, size_t ws_size,
                              hipStream_t stream) {
    const float* X     = (const float*)d_in[0];  // [B,N,128]
    const float* W     = (const float*)d_in[1];  // [128,256]
    const float* a_src = (const float*)d_in[2];  // [4,64,1]
    const float* a_dst = (const float*)d_in[3];  // [4,64,1]
    const float* bias  = (const float*)d_in[4];  // [256]
    float* out = (float*)d_out;                  // [B,N,256]

    char* ws = (char*)d_ws;
    size_t off = 0;
    __bf16* Xbf = (__bf16*)(ws + off); off += (size_t)GAT_M * GAT_IN * 2;        // 2 MB
    __bf16* WT  = (__bf16*)(ws + off); off += (size_t)GAT_HD * GAT_IN * 2;       // 64 KB
    float*  hF  = (float*) (ws + off); off += (size_t)GAT_M * GAT_HD * 4;        // 8 MB
    __bf16* hT  = (__bf16*)(ws + off); off += (size_t)GAT_B*GAT_H*GAT_D*GAT_N*2; // 4 MB
    float*  eS  = (float*) (ws + off); off += (size_t)GAT_B*GAT_H*GAT_N * 4;
    float*  eD  = (float*) (ws + off); off += (size_t)GAT_B*GAT_H*GAT_N * 4;
    float*  dM  = (float*) (ws + off); off += 256;

    gat_cvt_x <<<(GAT_M * GAT_IN) / 256, 256, 0, stream>>>(X, Xbf);
    gat_cvt_w <<<(GAT_HD * GAT_IN) / 256, 256, 0, stream>>>(W, WT);
    gat_proj  <<<(GAT_M / 16) * GAT_H / 8, 256, 0, stream>>>(Xbf, WT, hF, hT);
    gat_escore<<<(GAT_B * GAT_H * GAT_N) / 256, 256, 0, stream>>>(hF, a_src, a_dst, eS, eD);
    gat_dmax  <<<GAT_B * GAT_H, 256, 0, stream>>>(eD, dM);
    gat_flash <<<GAT_B * GAT_H * (GAT_N / 128), 256, 0, stream>>>(eS, eD, dM, hT, bias, out);
}
